// Lorenz96DBF_68642167324880
// MI455X (gfx1250) — compile-verified, matrix operand order checked by reading, not run on hardware
//
#include <hip/hip_runtime.h>
#include <math.h>

// ---------------------------------------------------------------------------
// Deep variational Kalman filter forward pass for MI455X (gfx1250, wave32).
// GEMMs via v_wmma_f32_16x16x32_bf16, BK=64 double-buffered LDS pipeline,
// async global->LDS staging (ASYNCcnt) when the toolchain exposes it.
// ---------------------------------------------------------------------------

typedef __attribute__((ext_vector_type(16))) __bf16         v16bf;
typedef __attribute__((ext_vector_type(8)))  float          v8f;
typedef __attribute__((ext_vector_type(8)))  unsigned short ushort8;
typedef __attribute__((ext_vector_type(4)))  unsigned short bfx4;   // avoid HIP's ushort4
typedef __attribute__((ext_vector_type(4)))  int            v4i;

#if defined(__has_builtin)
#  if __has_builtin(__builtin_amdgcn_global_load_async_to_lds_b128) && \
      __has_builtin(__builtin_amdgcn_s_wait_asynccnt)
#    define USE_ASYNC_LDS 1
#  endif
#endif
#ifndef USE_ASYNC_LDS
#  define USE_ASYNC_LDS 0
#endif

#if USE_ASYNC_LDS
typedef __attribute__((address_space(1))) v4i as1_v4i;   // global int4
typedef __attribute__((address_space(3))) v4i as3_v4i;   // LDS int4
#endif

#define LOG2PI 1.8378770664093453f
#define QVAL   0.13533528323661270f   // exp(LOG_Q) = exp(-2)

__device__ __forceinline__ unsigned short f2bf(float x) {
    unsigned int u = __float_as_uint(x);
    u += 0x7FFFu + ((u >> 16) & 1u);      // round to nearest even
    return (unsigned short)(u >> 16);
}

// branch-free tanh: lowers to v_exp_f32 + v_rcp_f32 (no OCML branches)
__device__ __forceinline__ float fast_tanh(float x) {
    const float ax = fabsf(x);
    const float e  = __expf(-2.f * ax);
    const float t  = (1.f - e) / (1.f + e);
    return copysignf(t, x);
}

union FragU { ushort8 u[2]; v16bf v; };

// ---------------------------------------------------------------------------
// Tiled WMMA GEMM:  Out = epilogue(A[MxK] * Bt^T + bias),  Bt is (N,K) bf16.
//   EPI 0: out_bf16 = bf16(tanh(acc + bias))
//   EPI 1: out_f32  = acc + bias
//   EPI 3: Gaussian NLL reduction vs target -> atomicAdd(accum[1])
// 256 threads (8 waves), tile 128x128x64, double-buffered LDS,
// wave grid 2(M) x 4(N); 16 WMMAs per wave per K-stage, 1 barrier per stage.
// ---------------------------------------------------------------------------
template<int EPI, bool AF32>
__global__ __launch_bounds__(256) void gemm_bf16_wmma(
    const void* __restrict__ Ap, const unsigned short* __restrict__ Bt,
    const float* __restrict__ bias, void* __restrict__ Outp,
    const float* __restrict__ target, const float* __restrict__ inv_var,
    const float* __restrict__ cterm, float* __restrict__ accum,
    int M, int N, int K)
{
    constexpr int BM = 128, BN = 128, BK = 64;
    constexpr int LDA = 72, LDB = 72;   // 144B pitch: 16B aligned, bank-friendly
    __shared__ __align__(16) unsigned short As[2][BM * LDA];
    __shared__ __align__(16) unsigned short Bs[2][BN * LDB];

    const int tid  = threadIdx.x;
    const int wave = tid >> 5, lane = tid & 31;
    const int half = lane >> 4, lr = lane & 15;
    const int m0 = blockIdx.x * BM, n0 = blockIdx.y * BN;
    const int wm = (wave & 1) * 64;     // 4 M-tiles per wave
    const int wn = (wave >> 1) * 32;    // 2 N-tiles per wave

    v8f acc[4][2] = {};

    const int kt_n = K / BK;

    // ---- tile stagers -----------------------------------------------------
    auto stageA = [&](int kt, int buf) {
        if constexpr (AF32) {
            const float* A = (const float*)Ap;
            #pragma unroll
            for (int it = 0; it < 8; ++it) {            // 2048 float4 slots
                int q = it * 256 + tid;
                int row = q >> 4, c4 = (q & 15) * 4;
                const float4 v = *(const float4*)(A + (size_t)(m0 + row) * K + kt * BK + c4);
                bfx4 p; p[0] = f2bf(v.x); p[1] = f2bf(v.y); p[2] = f2bf(v.z); p[3] = f2bf(v.w);
                *(bfx4*)&As[buf][row * LDA + c4] = p;   // 8B aligned
            }
        } else {
            const unsigned short* A = (const unsigned short*)Ap;
            #pragma unroll
            for (int it = 0; it < 4; ++it) {            // 1024 ushort8 slots
                int q = it * 256 + tid;
                int row = q >> 3, c8 = (q & 7) * 8;
                const unsigned short* gp = A + (size_t)(m0 + row) * K + kt * BK + c8;
                unsigned short* lp = &As[buf][row * LDA + c8];
#if USE_ASYNC_LDS
                __builtin_amdgcn_global_load_async_to_lds_b128(
                    (as1_v4i*)(void*)gp, (as3_v4i*)(void*)lp, 0, 0);
#else
                *(ushort8*)lp = *(const ushort8*)gp;
#endif
            }
        }
        if (kt + 1 < kt_n)   // warm WGP$/L2 one stage ahead of the copies
            __builtin_prefetch((const char*)Ap + ((size_t)(m0 + (tid >> 1)) * K +
                               (kt + 1) * BK + (tid & 1) * (BK / 2)) * (AF32 ? 4 : 2), 0, 3);
    };
    auto stageB = [&](int kt, int buf) {
        #pragma unroll
        for (int it = 0; it < 4; ++it) {                // 1024 ushort8 slots
            int q = it * 256 + tid;
            int n = q >> 3, k8 = (q & 7) * 8;
            const unsigned short* gp = Bt + (size_t)(n0 + n) * K + kt * BK + k8;
            unsigned short* lp = &Bs[buf][n * LDB + k8];
#if USE_ASYNC_LDS
            __builtin_amdgcn_global_load_async_to_lds_b128(
                (as1_v4i*)(void*)gp, (as3_v4i*)(void*)lp, 0, 0);
#else
            *(ushort8*)lp = *(const ushort8*)gp;        // 16B aligned, no scatter
#endif
        }
        if (kt + 1 < kt_n)
            __builtin_prefetch(Bt + (size_t)(n0 + (tid >> 1)) * K +
                               (kt + 1) * BK + (tid & 1) * (BK / 2), 0, 3);
    };

    // ---- software pipeline: stage(k+1) overlaps WMMA(k) -------------------
    stageA(0, 0);
    stageB(0, 0);
#if USE_ASYNC_LDS
    __builtin_amdgcn_s_wait_asynccnt(0);
#endif
    __syncthreads();

    int buf = 0;
    for (int kt = 0; kt < kt_n; ++kt) {
        const int nxt = buf ^ 1;
        if (kt + 1 < kt_n) { stageA(kt + 1, nxt); stageB(kt + 1, nxt); }

        // A 16x32 frag: lane half=0 holds K 0-7,16-23; half=1 holds K 8-15,24-31
        #pragma unroll
        for (int kk = 0; kk < 2; ++kk) {
            FragU a[4], b[2];
            #pragma unroll
            for (int i = 0; i < 4; ++i) {
                const unsigned short* p = &As[buf][(wm + i * 16 + lr) * LDA + kk * 32 + half * 8];
                a[i].u[0] = *(const ushort8*)p;
                a[i].u[1] = *(const ushort8*)(p + 16);
            }
            #pragma unroll
            for (int j = 0; j < 2; ++j) {
                const unsigned short* p = &Bs[buf][(wn + j * 16 + lr) * LDB + kk * 32 + half * 8];
                b[j].u[0] = *(const ushort8*)p;
                b[j].u[1] = *(const ushort8*)(p + 16);
            }
            #pragma unroll
            for (int i = 0; i < 4; ++i)
                #pragma unroll
                for (int j = 0; j < 2; ++j)
                    acc[i][j] = __builtin_amdgcn_wmma_f32_16x16x32_bf16(
                        false, a[i].v, false, b[j].v, (short)0, acc[i][j], false, false);
        }
#if USE_ASYNC_LDS
        __builtin_amdgcn_s_wait_asynccnt(0);   // drain stage(k+1) after overlap window
#endif
        __syncthreads();
        buf = nxt;
    }

    // ---- epilogue (C/D layout: lane -> N=lane&15; vgpr e -> M=e+8*(lane>=16)) ----
    float local = 0.0f;
    #pragma unroll
    for (int i = 0; i < 4; ++i) {
        #pragma unroll
        for (int j = 0; j < 2; ++j) {
            const int gn = n0 + wn + j * 16 + lr;
            const float bv = bias[gn];
            #pragma unroll
            for (int e = 0; e < 8; ++e) {
                const int gm = m0 + wm + i * 16 + half * 8 + e;
                const float v = acc[i][j][e] + bv;
                if constexpr (EPI == 0) {
                    ((unsigned short*)Outp)[(size_t)gm * N + gn] = f2bf(fast_tanh(v));
                } else if constexpr (EPI == 1) {
                    ((float*)Outp)[(size_t)gm * N + gn] = v;
                } else { // EPI == 3: Gaussian NLL term
                    const float d = target[(size_t)gm * N + gn] - v;
                    local += cterm[gn] + 0.5f * d * d * inv_var[gn];
                }
            }
        }
    }
    if constexpr (EPI == 3) {
        __syncthreads();
        float* red = (float*)As;
        red[tid] = local;
        __syncthreads();
        for (int s = 128; s > 0; s >>= 1) {
            if (tid < s) red[tid] += red[tid + s];
            __syncthreads();
        }
        if (tid == 0) atomicAdd(accum + 1, red[0]);
    }
}

// ---------------------------------------------------------------------------
// Sequential 2x2 Kalman scan: one thread per (batch, latent-block) chain.
// 64*256 = 16384 chains x 200 steps. Emits z (bf16 packed) and the KL sum.
// ---------------------------------------------------------------------------
__global__ __launch_bounds__(256) void kalman_scan(
    const float* __restrict__ enc, const float* __restrict__ eps,
    const float* __restrict__ lambdas, unsigned int* __restrict__ zb,
    float* __restrict__ accum)
{
    const int tid = threadIdx.x;
    const int idx = blockIdx.x * 256 + tid;
    const int bb = idx >> 8;            // batch
    const int z  = idx & 255;           // 2x2 block index

    const float lam0 = lambdas[2 * z], lam1 = lambdas[2 * z + 1];
    const float r  = 1.0f / (1.0f + __expf(-lam0));
    const float cs = __cosf(lam1), sn = __sinf(lam1);
    const float a00 = r * cs, a01 = -r * sn, a10 = r * sn, a11 = r * cs;

    float mu0 = 0.f, mu1 = 0.f;
    float sp00 = 10.f, sp01 = 0.f, sp11 = 10.f;   // INIT_COV * I
    float kl = 0.f;

    for (int t = 0; t < 200; ++t) {
        const size_t rowe = (size_t)(bb * 200 + t) * 1024;
        const float2 fv = *(const float2*)(enc + rowe + 2 * z);
        const float2 gv = *(const float2*)(enc + rowe + 512 + 2 * z);
        const float g0 = 100.f * fast_tanh(gv.x * gv.x * 0.01f);
        const float g1 = 100.f * fast_tanh(gv.y * gv.y * 0.01f);
        const float gf0 = g0 * fv.x, gf1 = g1 * fv.y;

        // s_pred^-1
        const float det_sp = sp00 * sp11 - sp01 * sp01;
        const float ids = 1.f / det_sp;
        const float i00 = sp11 * ids, i01 = -sp01 * ids, i11 = sp00 * ids;

        // s_filt = (s_pred^-1 + diag(g))^-1  (symmetric by construction)
        const float m00 = i00 + g0, m01 = i01, m11 = i11 + g1;
        const float det_m = m00 * m11 - m01 * m01;
        const float idm = 1.f / det_m;
        const float f00 = m11 * idm, f01 = -m01 * idm, f11 = m00 * idm;

        // mu_filt = s_filt * s_pred^-1 * mu_pred + g*f
        const float t0 = i00 * mu0 + i01 * mu1;
        const float t1 = i01 * mu0 + i11 * mu1;
        const float mf0 = f00 * t0 + f01 * t1 + gf0;
        const float mf1 = f01 * t0 + f11 * t1 + gf1;

        // KL(q_filt || p_pred); log det s_filt = -log det_m
        const float tr   = i00 * f00 + 2.f * i01 * f01 + i11 * f11;
        const float d0 = mu0 - mf0, d1 = mu1 - mf1;
        const float quad = i00 * d0 * d0 + 2.f * i01 * d0 * d1 + i11 * d1 * d1;
        kl += 0.5f * (tr + quad - 2.f + __logf(det_sp) + __logf(det_m));

        // Cholesky sample z = mu_filt + L*eps
        const float l00 = sqrtf(f00);
        const float l10 = f01 / l00;
        const float l11 = sqrtf(fmaxf(f11 - l10 * l10, 0.f));
        const float2 ev = *(const float2*)(eps + ((size_t)(bb * 200 + t) * 256 + z) * 2);
        const float z0 = mf0 + l00 * ev.x;
        const float z1 = mf1 + l10 * ev.x + l11 * ev.y;
        zb[(size_t)(bb * 200 + t) * 256 + z] =
            (unsigned int)f2bf(z0) | ((unsigned int)f2bf(z1) << 16);

        // propagate: mu' = A mu_f ; S' = A S_f A^T + exp(-2) I
        const float nm0 = a00 * mf0 + a01 * mf1;
        const float nm1 = a10 * mf0 + a11 * mf1;
        const float b00 = a00 * f00 + a01 * f01, b01 = a00 * f01 + a01 * f11;
        const float b10 = a10 * f00 + a11 * f01, b11 = a10 * f01 + a11 * f11;
        sp00 = b00 * a00 + b01 * a01 + QVAL;
        sp01 = b00 * a10 + b01 * a11;
        sp11 = b10 * a10 + b11 * a11 + QVAL;
        mu0 = nm0; mu1 = nm1;
    }

    __shared__ float red[256];
    red[tid] = kl;
    __syncthreads();
    for (int s = 128; s > 0; s >>= 1) {
        if (tid < s) red[tid] += red[tid + s];
        __syncthreads();
    }
    if (tid == 0) atomicAdd(accum, red[0]);
}

// ---------------------------------------------------------------------------
// Small helpers
// ---------------------------------------------------------------------------
__global__ void init_accum(float* a) { a[0] = 0.f; a[1] = 0.f; }

// convert (K,N) f32 row-major -> (N,K) bf16 row-major (i.e. transposed)
__global__ void cvt_transpose_bf16(const float* __restrict__ s,
                                   unsigned short* __restrict__ d, int K, int N) {
    const int total = K * N;
    for (int i = blockIdx.x * blockDim.x + threadIdx.x; i < total;
         i += gridDim.x * blockDim.x) {
        const int k = i / N, n = i - k * N;
        d[(size_t)n * K + k] = f2bf(s[i]);
    }
}

__global__ void col_prep(const float* __restrict__ logR,
                         float* __restrict__ inv_var, float* __restrict__ cterm) {
    const int n = threadIdx.x;              // 256 columns
    const float lr = logR[n];
    inv_var[n] = __expf(-2.f * lr);
    cterm[n]   = 0.5f * (LOG2PI + 2.f * lr);
}

__global__ void finalize(const float* __restrict__ accum, float* __restrict__ out) {
    const float kl  = accum[0] / 64.f;      // / B
    const float nll = accum[1] / 64.f;
    out[0] = kl + nll;
    out[1] = kl;
    out[2] = nll;
}

// ---------------------------------------------------------------------------
extern "C" void kernel_launch(void* const* d_in, const int* in_sizes, int n_in,
                              void* d_out, int out_size, void* d_ws, size_t ws_size,
                              hipStream_t stream)
{
    const float* obs     = (const float*)d_in[0];
    const float* target  = (const float*)d_in[1];
    const float* lambdas = (const float*)d_in[2];
    const float* logR    = (const float*)d_in[3];
    const float* eps     = (const float*)d_in[4];
    const float* W1 = (const float*)d_in[5];
    const float* b1 = (const float*)d_in[6];
    const float* W2 = (const float*)d_in[7];
    const float* b2 = (const float*)d_in[8];
    const float* V1 = (const float*)d_in[9];
    const float* c1 = (const float*)d_in[10];
    const float* V2 = (const float*)d_in[11];
    const float* c2 = (const float*)d_in[12];

    char* ws = (char*)d_ws;
    float*          accum = (float*)ws;                         // 2 floats
    unsigned short* W1t   = (unsigned short*)(ws + 256);        // (1024,256) bf16
    unsigned short* W2t   = (unsigned short*)(ws + 524544);     // (1024,1024) bf16
    unsigned short* V1t   = (unsigned short*)(ws + 2621696);    // (1024,512) bf16
    unsigned short* V2t   = (unsigned short*)(ws + 3670272);    // (256,1024) bf16
    float*          ivar  = (float*)(ws + 4194560);             // 256 f32
    float*          cterm = (float*)(ws + 4195584);             // 256 f32
    unsigned short* hb    = (unsigned short*)(ws + 4196608);    // 12800x1024 bf16
    float*          enc   = (float*)(ws + 30411008);            // 12800x1024 f32
    unsigned int*   zb    = (unsigned int*)(ws + 82839808);     // 12800x256 packed bf16x2
    unsigned short* h2b   = (unsigned short*)(ws + 95947008);   // 12800x1024 bf16
    // total workspace footprint ~122.2 MB

    init_accum<<<1, 1, 0, stream>>>(accum);
    cvt_transpose_bf16<<<256, 256, 0, stream>>>(W1, W1t, 256, 1024);
    cvt_transpose_bf16<<<256, 256, 0, stream>>>(W2, W2t, 1024, 1024);
    cvt_transpose_bf16<<<256, 256, 0, stream>>>(V1, V1t, 512, 1024);
    cvt_transpose_bf16<<<256, 256, 0, stream>>>(V2, V2t, 1024, 256);
    col_prep<<<1, 256, 0, stream>>>(logR, ivar, cterm);

    const dim3 gWide(100, 8);   // 12800/128 x 1024/128
    const dim3 gOut (100, 2);   // 12800/128 x  256/128

    // encoder: h = tanh(obs @ W1 + b1)        (A is f32)
    gemm_bf16_wmma<0, true ><<<gWide, 256, 0, stream>>>(
        obs, W1t, b1, hb, nullptr, nullptr, nullptr, accum, 12800, 1024, 256);
    // enc = h @ W2 + b2                        (f32 out, consumed by scan)
    gemm_bf16_wmma<1, false><<<gWide, 256, 0, stream>>>(
        hb, W2t, b2, enc, nullptr, nullptr, nullptr, accum, 12800, 1024, 1024);
    // Kalman scan + KL + sampling (z -> packed bf16)
    kalman_scan<<<64, 256, 0, stream>>>(enc, eps, lambdas, zb, accum);
    // decoder: h2 = tanh(z @ V1 + c1)
    gemm_bf16_wmma<0, false><<<gWide, 256, 0, stream>>>(
        zb, V1t, c1, h2b, nullptr, nullptr, nullptr, accum, 12800, 1024, 512);
    // recon + fused Gaussian NLL reduction
    gemm_bf16_wmma<3, false><<<gOut, 256, 0, stream>>>(
        h2b, V2t, c2, nullptr, target, ivar, cterm, accum, 12800, 256, 1024);

    finalize<<<1, 1, 0, stream>>>(accum, (float*)d_out);
}
